// GRUCell1_8194797601187
// MI455X (gfx1250) — matrix operand order repaired and checked by reading, not compiled, not run on hardware
//
#include <hip/hip_runtime.h>
#include <hip/hip_bf16.h>

// CDNA5 / gfx1250: wave32, WMMA 16x16x32 bf16 with f32 accumulation.
typedef __attribute__((ext_vector_type(16))) __bf16 v16bf;
typedef __attribute__((ext_vector_type(8)))  float  v8f;

#if defined(__has_builtin)
# if __has_builtin(__builtin_amdgcn_tanhf)
#  define GRU_TANH_NATIVE 1
# endif
# if __has_builtin(__builtin_amdgcn_cvt_pk_bf16_f32)
#  define GRU_PK_BF16 1
# endif
#endif

union Frag {
    v16bf bf;
    unsigned short u[16];
    unsigned d[8];
    uint4 q[2];
};

__device__ __forceinline__ unsigned short f2bf(float f) {
    unsigned u = __float_as_uint(f);
    return (unsigned short)((u + 0x7FFFu + ((u >> 16) & 1u)) >> 16); // RNE
}
__device__ __forceinline__ unsigned pack2(float a, float b) {
#if defined(GRU_PK_BF16)
    auto r = __builtin_amdgcn_cvt_pk_bf16_f32(a, b);
    unsigned u;
    __builtin_memcpy(&u, &r, 4);
    return u;
#else
    return (unsigned)f2bf(a) | ((unsigned)f2bf(b) << 16);
#endif
}
__device__ __forceinline__ float fast_tanh(float x) {
#if defined(GRU_TANH_NATIVE)
    return __builtin_amdgcn_tanhf(x);          // v_tanh_f32 (CDNA5 TRANS op)
#else
    float e = __expf(2.0f * x);                // division-free fallback
    return 1.0f - 2.0f * __builtin_amdgcn_rcpf(e + 1.0f);
#endif
}

// P=2048, K=8, T=32, C=16384, F=128.  16 paths per workgroup, 8 waves.
__global__ __launch_bounds__(256, 1)
void gru_path_kernel(const float* __restrict__ h0,
                     const float* __restrict__ cf,    // [C,T,F]
                     const float* __restrict__ wih,   // [384,128]
                     const float* __restrict__ whh,   // [384,128]
                     const float* __restrict__ bih,   // [384]
                     const float* __restrict__ bhh,   // [384]
                     const long long* __restrict__ idx, // [P,8]
                     float* __restrict__ out)         // [P,128]
{
    __shared__ __align__(16) unsigned short xbuf[3][16 * 128]; // bf16 x tiles
    __shared__ __align__(16) unsigned short hbuf[2][16 * 128]; // bf16 h tiles
    __shared__ unsigned long long baseTab[16][8];              // gather bases (elems)

    const int tid  = threadIdx.x;
    const int lane = tid & 31;
    const int wv   = tid >> 5;      // wave 0..7 -> gate column tile 16*wv
    const int g    = lane >> 4;     // lane half
    const int cl   = lane & 15;     // column / row-within-half
    const int p0   = blockIdx.x * 16;

    // ---- gather base table: idx * T * F ----
    if (tid < 128) {
        int m = tid >> 3, k = tid & 7;
        baseTab[m][k] = (unsigned long long)idx[(size_t)(p0 + m) * 8 + k] * 4096ull;
    }

    // ---- resident weight B-fragments (contiguous K order, same as A: elem e -> K = 32c + 16g + e) ----
    Frag wih_f[3][4], whh_f[3][4];
    const int nbase[3] = {16 * wv, 128 + 16 * wv, 256 + 16 * wv};
#pragma unroll
    for (int t = 0; t < 3; ++t)
#pragma unroll
        for (int c = 0; c < 4; ++c) {
            const float* pi = wih + (size_t)(nbase[t] + cl) * 128 + 32 * c + 16 * g;
            const float* ph = whh + (size_t)(nbase[t] + cl) * 128 + 32 * c + 16 * g;
#pragma unroll
            for (int e = 0; e < 8; ++e) {
                wih_f[t][c].d[e] = pack2(pi[2 * e], pi[2 * e + 1]);
                whh_f[t][c].d[e] = pack2(ph[2 * e], ph[2 * e + 1]);
            }
        }

    // ---- biases folded into the scalar gate stage (pre-scaled for sigmoid-as-tanh) ----
    const float bR2 = 0.5f * (bih[nbase[0] + cl] + bhh[nbase[0] + cl]);
    const float bZ2 = 0.5f * (bih[nbase[1] + cl] + bhh[nbase[1] + cl]);
    const float bXN = bih[nbase[2] + cl];
    const float bHN = bhh[nbase[2] + cl];

    // ---- h0 into D-fragment registers (elem r -> row r+8g, col 16wv+cl) ----
    float h_old[8];
#pragma unroll
    for (int r = 0; r < 8; ++r)
        h_old[r] = h0[(size_t)(p0 + r + 8 * g) * 128 + 16 * wv + cl];
#pragma unroll
    for (int r = 0; r < 8; ++r)
        hbuf[0][(r + 8 * g) * 128 + 16 * wv + cl] = f2bf(h_old[r]);

    __syncthreads(); // baseTab + hbuf[0] visible

    // ---- cooperative x loader identity: 8 consecutive floats per thread ----
    const int m_ld = tid >> 4;
    const int col0 = (tid & 15) * 8;

    // prologue: stage x(0), x(1); issue pending loads for x(2)
#pragma unroll
    for (int sp = 0; sp < 2; ++sp) {
        const float* src = cf + baseTab[m_ld][0] + (size_t)sp * 128 + col0;
        uint4 o;
        o.x = pack2(src[0], src[1]); o.y = pack2(src[2], src[3]);
        o.z = pack2(src[4], src[5]); o.w = pack2(src[6], src[7]);
        *(uint4*)&xbuf[sp][m_ld * 128 + col0] = o;
    }
    float4 pfa, pfb;
    {
        const float4* s2 = (const float4*)(cf + baseTab[m_ld][0] + (size_t)2 * 128 + col0);
        pfa = s2[0]; pfb = s2[1];
    }
    __syncthreads();

    // loop-invariant zero accumulator seed (keeps per-step v_movs off the critical path)
    v8f zseed;
#pragma unroll
    for (int i = 0; i < 8; ++i) zseed[i] = 0.0f;

    // ================= recurrence: 256 serial steps =================
    for (int s = 0; s < 256; ++s) {
        const unsigned short* xb = xbuf[s % 3];
        const unsigned short* hb = hbuf[s & 1];

        // A-fragments (contiguous: half g reads K = 32c+16g .. 32c+16g+15)
        Frag xa[4], ha[4];
#pragma unroll
        for (int c = 0; c < 4; ++c) {
            const int base = cl * 128 + 32 * c + 16 * g;
            xa[c].q[0] = *(const uint4*)&xb[base];
            xa[c].q[1] = *(const uint4*)&xb[base + 8];
            ha[c].q[0] = *(const uint4*)&hb[base];
            ha[c].q[1] = *(const uint4*)&hb[base + 8];
        }

        v8f accR = zseed, accZ = zseed, accXN = zseed, accHN = zseed;
#pragma unroll
        for (int c = 0; c < 4; ++c) {
            accR  = __builtin_amdgcn_wmma_f32_16x16x32_bf16(false, xa[c].bf, false, wih_f[0][c].bf, (short)0, accR,  false, false);
            accR  = __builtin_amdgcn_wmma_f32_16x16x32_bf16(false, ha[c].bf, false, whh_f[0][c].bf, (short)0, accR,  false, false);
            accZ  = __builtin_amdgcn_wmma_f32_16x16x32_bf16(false, xa[c].bf, false, wih_f[1][c].bf, (short)0, accZ,  false, false);
            accZ  = __builtin_amdgcn_wmma_f32_16x16x32_bf16(false, ha[c].bf, false, whh_f[1][c].bf, (short)0, accZ,  false, false);
            accXN = __builtin_amdgcn_wmma_f32_16x16x32_bf16(false, xa[c].bf, false, wih_f[2][c].bf, (short)0, accXN, false, false);
            accHN = __builtin_amdgcn_wmma_f32_16x16x32_bf16(false, ha[c].bf, false, whh_f[2][c].bf, (short)0, accHN, false, false);
        }

        // gates: sigmoid(x) = 0.5*tanh(0.5x)+0.5 (no division); h = fma(z, h-n, n)
#pragma unroll
        for (int r = 0; r < 8; ++r) {
            float rg = fmaf(fast_tanh(fmaf(0.5f, accR[r], bR2)), 0.5f, 0.5f);
            float zg = fmaf(fast_tanh(fmaf(0.5f, accZ[r], bZ2)), 0.5f, 0.5f);
            float ng = fast_tanh(fmaf(rg, accHN[r] + bHN, accXN[r] + bXN));
            h_old[r] = fmaf(zg, h_old[r] - ng, ng);
        }
        // publish new h (bf16, double buffer)
        unsigned short* hbn = hbuf[(s + 1) & 1];
#pragma unroll
        for (int r = 0; r < 8; ++r)
            hbn[(r + 8 * g) * 128 + 16 * wv + cl] = f2bf(h_old[r]);

        // x pipeline: store s+2 (loads issued last iter), issue loads for s+3
        if (s + 2 < 256) {
            uint4 o;
            o.x = pack2(pfa.x, pfa.y); o.y = pack2(pfa.z, pfa.w);
            o.z = pack2(pfb.x, pfb.y); o.w = pack2(pfb.z, pfb.w);
            *(uint4*)&xbuf[(s + 2) % 3][m_ld * 128 + col0] = o;
            if (s + 3 < 256) {
                const int k = (s + 3) >> 5, t = (s + 3) & 31;
                const float4* src = (const float4*)(cf + baseTab[m_ld][k] + (size_t)t * 128 + col0);
                pfa = src[0]; pfb = src[1];
            }
        }
        __syncthreads();
    }

    // ---- final hidden state ----
#pragma unroll
    for (int r = 0; r < 8; ++r)
        out[(size_t)(p0 + r + 8 * g) * 128 + 16 * wv + cl] = h_old[r];
}

extern "C" void kernel_launch(void* const* d_in, const int* in_sizes, int n_in,
                              void* d_out, int out_size, void* d_ws, size_t ws_size,
                              hipStream_t stream) {
    const float*     h0  = (const float*)d_in[0];
    const float*     cf  = (const float*)d_in[1];
    const float*     wih = (const float*)d_in[2];
    const float*     whh = (const float*)d_in[3];
    const float*     bih = (const float*)d_in[4];
    const float*     bhh = (const float*)d_in[5];
    const long long* idx = (const long long*)d_in[6];
    float*           out = (float*)d_out;
    (void)in_sizes; (void)n_in; (void)out_size; (void)d_ws; (void)ws_size;

    gru_path_kernel<<<2048 / 16, 256, 0, stream>>>(h0, cf, wih, whh, bih, bhh, idx, out);
}
